// GroupAttention_18829136626079
// MI455X (gfx1250) — matrix operand order repaired and checked by reading, not compiled, not run on hardware
//
#include <hip/hip_runtime.h>
#include <stdint.h>

#define DIMX 1024
#define HEADS 16
#define DHEAD 64
#define INNERX 1024
#define BATCH 4
#define TDIM 16
#define NDIM 256
#define ROWS (BATCH * TDIM * NDIM)     /* 16384 */
#define GROUPS (BATCH * HEADS * TDIM)  /* 1024  */

typedef unsigned short ushort_t;
typedef __attribute__((ext_vector_type(16))) __bf16 v16bf;
typedef __attribute__((ext_vector_type(8)))  __bf16 v8bf;
typedef __attribute__((ext_vector_type(8)))  float  v8f;

static __device__ __forceinline__ ushort_t f2bf(float f) {
    unsigned int u = __float_as_uint(f);
    u += 0x7FFFu + ((u >> 16) & 1u);   // round-to-nearest-even
    return (ushort_t)(u >> 16);
}

static __device__ __forceinline__ v8f zerov() {
    v8f z = {0.f, 0.f, 0.f, 0.f, 0.f, 0.f, 0.f, 0.f};
    return z;
}

static __device__ __forceinline__ v8f wmma_bf16(v16bf a, v16bf b, v8f c) {
    // D = A(16x32 bf16) * B(32x16 bf16) + C(16x16 f32)
    return __builtin_amdgcn_wmma_f32_16x16x32_bf16(
        /*neg_a=*/false, a, /*neg_b=*/false, b,
        /*c_mod=*/(short)0, c, /*reuse_a=*/false, /*reuse_b=*/false);
}

static __device__ __forceinline__ v16bf cat16(v8bf lo, v8bf hi) {
    v16bf r;
#pragma unroll
    for (int i = 0; i < 8; ++i) { r[i] = lo[i]; r[i + 8] = hi[i]; }
    return r;
}

// ---------------------------------------------------------------------------
// Kernel 1: LayerNorm (fp32 in) -> bf16 xn.  One block (256 threads) per row.
// ---------------------------------------------------------------------------
__global__ void __launch_bounds__(256)
ln_kernel(const float* __restrict__ x, const float* __restrict__ gamma,
          const float* __restrict__ beta, ushort_t* __restrict__ xn) {
    const int row = blockIdx.x;
    const int tid = threadIdx.x;
    const float4 v = ((const float4*)(x + (size_t)row * DIMX))[tid];
    float s1 = v.x + v.y + v.z + v.w;
    float s2 = v.x * v.x + v.y * v.y + v.z * v.z + v.w * v.w;
#pragma unroll
    for (int off = 16; off; off >>= 1) {
        s1 += __shfl_xor(s1, off, 32);
        s2 += __shfl_xor(s2, off, 32);
    }
    __shared__ float red1[8], red2[8];
    const int wv = tid >> 5;
    if ((tid & 31) == 0) { red1[wv] = s1; red2[wv] = s2; }
    __syncthreads();
    float t1 = 0.f, t2 = 0.f;
#pragma unroll
    for (int i = 0; i < 8; ++i) { t1 += red1[i]; t2 += red2[i]; }
    const float mean = t1 * (1.f / DIMX);
    const float var  = t2 * (1.f / DIMX) - mean * mean;
    const float rstd = rsqrtf(var + 1e-5f);
    const float4 g = ((const float4*)gamma)[tid];
    const float4 b = ((const float4*)beta)[tid];
    ushort_t* o = xn + (size_t)row * DIMX + tid * 4;
    o[0] = f2bf((v.x - mean) * rstd * g.x + b.x);
    o[1] = f2bf((v.y - mean) * rstd * g.y + b.y);
    o[2] = f2bf((v.z - mean) * rstd * g.z + b.z);
    o[3] = f2bf((v.w - mean) * rstd * g.w + b.w);
}

// ---------------------------------------------------------------------------
// Kernel 2: transpose + cast weights: src[K][N] fp32 -> dst[N][K] bf16.
// ---------------------------------------------------------------------------
__global__ void __launch_bounds__(256)
tcast_kernel(const float* __restrict__ src, ushort_t* __restrict__ dst,
             int K, int N) {
    const size_t total = (size_t)K * N;
    for (size_t i = (size_t)blockIdx.x * blockDim.x + threadIdx.x; i < total;
         i += (size_t)gridDim.x * blockDim.x) {
        const size_t c = i / (size_t)K, k = i % (size_t)K;
        dst[i] = f2bf(src[k * (size_t)N + c]);
    }
}

// ---------------------------------------------------------------------------
// Kernel 3/5: WMMA GEMM.  A[16384][1024] bf16 row-major, WT[1024+][1024] bf16
// (row = output column).  Each wave computes a 64x64 output block:
// 16 accumulators (128 VGPRs), 16 WMMAs per K=32 step against 16 b128 loads
// (1 load per WMMA).  All fragment addresses fold into instruction offsets
// from two base pointers (row strides are 32 KB < 24-bit ioffset range).
// mode 0: write q  = y*0.125 as [g][n][d] bf16
// mode 1: write k  as [g][n][d] bf16
// mode 2: write vT as [g][d][n] bf16
// mode 3: write fp32 row-major [r][c] (final output)
// ---------------------------------------------------------------------------
__global__ void __launch_bounds__(256)
gemm_kernel(const ushort_t* __restrict__ A, const ushort_t* __restrict__ WT,
            void* __restrict__ outp, int mode) {
    const int K = DIMX;
    const int w    = blockIdx.x * 8 + (threadIdx.x >> 5);
    const int lane = threadIdx.x & 31;
    const int lrow = lane & 15, lhi = lane >> 4;
    const int mb = (w >> 4) << 6;   // 256 M-blocks of 64
    const int nb = (w & 15) << 6;   // 16  N-blocks of 64

    const ushort_t* aBase = A  + (size_t)(mb + lrow) * K + lhi * 16;
    const ushort_t* bBase = WT + (size_t)(nb + lrow) * K + lhi * 16;

    v8f acc[4][4];
#pragma unroll
    for (int i = 0; i < 4; ++i)
#pragma unroll
        for (int j = 0; j < 4; ++j) acc[i][j] = zerov();

    for (int k = 0; k < K; k += 32) {
        v16bf af[4], bf[4];
#pragma unroll
        for (int i = 0; i < 4; ++i) {
            const ushort_t* p = aBase + (size_t)i * 16 * K + k;
            af[i] = cat16(*(const v8bf*)p, *(const v8bf*)(p + 8));
        }
#pragma unroll
        for (int j = 0; j < 4; ++j) {
            bf[j] = *(const v16bf*)(bBase + (size_t)j * 16 * K + k);
        }
#pragma unroll
        for (int i = 0; i < 4; ++i)
#pragma unroll
            for (int j = 0; j < 4; ++j)
                acc[i][j] = wmma_bf16(af[i], bf[j], acc[i][j]);
    }

    const float scale = (mode == 0) ? 0.125f : 1.0f;
#pragma unroll
    for (int tm = 0; tm < 4; ++tm)
#pragma unroll
    for (int tn = 0; tn < 4; ++tn)
#pragma unroll
    for (int vr = 0; vr < 8; ++vr) {
        const int r  = mb + tm * 16 + vr + lhi * 8;
        const int cc = nb + tn * 16 + lrow;
        const float val = acc[tm][tn][vr] * scale;
        if (mode == 3) {
            ((float*)outp)[(size_t)r * DIMX + cc] = val;
        } else {
            const int b = r >> 12, t = (r >> 8) & 15, n = r & 255;
            const int h = cc >> 6, d = cc & 63;
            size_t idx;
            if (mode == 2)
                idx = ((((size_t)(b * HEADS + h) * TDIM + t) * DHEAD + d) * NDIM + n);
            else
                idx = ((((size_t)(b * HEADS + h) * TDIM + t) * NDIM + n) * DHEAD + d);
            ((ushort_t*)outp)[idx] = f2bf(val);
        }
    }
}

// ---------------------------------------------------------------------------
// Kernel 4: attention per (b,h,t) group. 8 waves/block; each wave owns two
// 16-row strips: QK^T (16 WMMA tiles wide) -> register softmax -> LDS bf16
// probs -> attn@V (WMMA) -> scatter to [b, t*n, h*d] bf16.
// ---------------------------------------------------------------------------
__global__ void __launch_bounds__(256)
attn_kernel(const ushort_t* __restrict__ q, const ushort_t* __restrict__ kmat,
            const ushort_t* __restrict__ vT, ushort_t* __restrict__ aout) {
    __shared__ ushort_t smem[8 * 16 * 256];   // 64 KB: one 16x256 strip / wave
    const int g = blockIdx.x;
    const int b = g >> 8, h = (g >> 4) & 15, t = g & 15;
    const int wv   = threadIdx.x >> 5;
    const int lane = threadIdx.x & 31;
    const int lrow = lane & 15, lhi = lane >> 4;
    const ushort_t* qg = q    + (size_t)g * (NDIM * DHEAD);
    const ushort_t* kg = kmat + (size_t)g * (NDIM * DHEAD);
    const ushort_t* vg = vT   + (size_t)g * (DHEAD * NDIM);
    ushort_t* sw = smem + wv * (16 * 256);

    for (int sm = 0; sm < 2; ++sm) {
        const int m0 = (wv + sm * 8) * 16;

        // ---- Phase 1: sim strip = q[m0:m0+16] @ k^T  (f32 accum) ----
        v8f c[16];
#pragma unroll
        for (int j = 0; j < 16; ++j) c[j] = zerov();
#pragma unroll
        for (int kb = 0; kb < DHEAD; kb += 32) {
            const ushort_t* ap = qg + (size_t)(m0 + lrow) * DHEAD + kb + lhi * 16;
            v16bf a = cat16(*(const v8bf*)ap, *(const v8bf*)(ap + 8));
#pragma unroll
            for (int j = 0; j < 16; ++j) {
                v16bf bb = *(const v16bf*)(kg + (size_t)(j * 16 + lrow) * DHEAD + kb + lhi * 16);
                c[j] = wmma_bf16(a, bb, c[j]);
            }
        }

        // ---- Phase 2: softmax rows (each half-wave owns a row per vr) ----
#pragma unroll
        for (int vr = 0; vr < 8; ++vr) {
            float m = c[0][vr];
#pragma unroll
            for (int j = 1; j < 16; ++j) m = fmaxf(m, c[j][vr]);
#pragma unroll
            for (int off = 8; off; off >>= 1) m = fmaxf(m, __shfl_xor(m, off, 32));
            float s = 0.f;
#pragma unroll
            for (int j = 0; j < 16; ++j) {
                float e = __expf(c[j][vr] - m);
                c[j][vr] = e;
                s += e;
            }
#pragma unroll
            for (int off = 8; off; off >>= 1) s += __shfl_xor(s, off, 32);
            const float inv = 1.0f / s;
#pragma unroll
            for (int j = 0; j < 16; ++j)
                sw[(vr + lhi * 8) * 256 + j * 16 + lrow] = f2bf(c[j][vr] * inv);
        }
        // LDS ops from the same wave are in-order (DScnt), no barrier needed:
        // each wave reads back only its own strip.

        // ---- Phase 3: out strip = attn @ v  (v stored transposed [d][n]) ----
        v8f o[4];
#pragma unroll
        for (int d = 0; d < 4; ++d) o[d] = zerov();
#pragma unroll
        for (int kb = 0; kb < NDIM; kb += 32) {
            const ushort_t* ap = sw + lrow * 256 + kb + lhi * 16;
            v16bf a = cat16(*(const v8bf*)ap, *(const v8bf*)(ap + 8));
#pragma unroll
            for (int d = 0; d < 4; ++d) {
                v16bf bb = *(const v16bf*)(vg + (size_t)(d * 16 + lrow) * NDIM + kb + lhi * 16);
                o[d] = wmma_bf16(a, bb, o[d]);
            }
        }

        // ---- Phase 4: scatter to [b, t*256+n, h*64+d] bf16 ----
#pragma unroll
        for (int dt = 0; dt < 4; ++dt)
#pragma unroll
        for (int vr = 0; vr < 8; ++vr) {
            const int n  = m0 + vr + lhi * 8;
            const int r  = b * (TDIM * NDIM) + t * NDIM + n;
            const int cc = h * DHEAD + dt * 16 + lrow;
            aout[(size_t)r * INNERX + cc] = f2bf(o[dt][vr]);
        }
    }
}

// ---------------------------------------------------------------------------
extern "C" void kernel_launch(void* const* d_in, const int* in_sizes, int n_in,
                              void* d_out, int out_size, void* d_ws, size_t ws_size,
                              hipStream_t stream) {
    (void)in_sizes; (void)n_in; (void)out_size; (void)ws_size;
    const float* x     = (const float*)d_in[0];
    const float* gamma = (const float*)d_in[1];
    const float* beta  = (const float*)d_in[2];
    const float* Wq    = (const float*)d_in[3];
    const float* Wkv   = (const float*)d_in[4];
    const float* Wout  = (const float*)d_in[5];

    char* ws = (char*)d_ws;
    size_t off = 0;
    auto alloc = [&](size_t bytes) -> char* {
        char* p = ws + off;
        off += (bytes + 255) & ~(size_t)255;
        return p;
    };
    ushort_t* xn    = (ushort_t*)alloc((size_t)ROWS * DIMX * 2);        // 32 MB
    ushort_t* WqT   = (ushort_t*)alloc((size_t)INNERX * DIMX * 2);      //  2 MB
    ushort_t* WkvT  = (ushort_t*)alloc((size_t)2 * INNERX * DIMX * 2);  //  4 MB
    ushort_t* WoutT = (ushort_t*)alloc((size_t)DIMX * INNERX * 2);      //  2 MB
    ushort_t* qb    = (ushort_t*)alloc((size_t)ROWS * INNERX * 2);      // 32 MB
    ushort_t* kbuf  = (ushort_t*)alloc((size_t)ROWS * INNERX * 2);      // 32 MB
    ushort_t* vTb   = (ushort_t*)alloc((size_t)ROWS * INNERX * 2);      // 32 MB
    ushort_t* ao    = (ushort_t*)alloc((size_t)ROWS * INNERX * 2);      // 32 MB

    ln_kernel<<<ROWS, 256, 0, stream>>>(x, gamma, beta, xn);
    tcast_kernel<<<1024, 256, 0, stream>>>(Wq, WqT, DIMX, INNERX);
    tcast_kernel<<<2048, 256, 0, stream>>>(Wkv, WkvT, DIMX, 2 * INNERX);
    tcast_kernel<<<1024, 256, 0, stream>>>(Wout, WoutT, INNERX, DIMX);

    const int gemm_blocks = (ROWS / 64) * (INNERX / 64) / 8;  // 512
    gemm_kernel<<<gemm_blocks, 256, 0, stream>>>(xn, WqT, qb, 0);
    gemm_kernel<<<gemm_blocks, 256, 0, stream>>>(xn, WkvT, kbuf, 1);
    gemm_kernel<<<gemm_blocks, 256, 0, stream>>>(xn, WkvT + (size_t)INNERX * DIMX, vTb, 2);

    attn_kernel<<<GROUPS, 256, 0, stream>>>(qb, kbuf, vTb, ao);

    gemm_kernel<<<gemm_blocks, 256, 0, stream>>>(ao, WoutT, d_out, 3);
}
